// Encoder_Block_38860864094420
// MI455X (gfx1250) — compile-verified, matrix operand order checked by reading, not run on hardware
//
#include <hip/hip_runtime.h>
#include <stdint.h>

// ---------------------------------------------------------------------------
// Problem dims (compile-time constants from the reference)
// ---------------------------------------------------------------------------
#define S_  4096
#define E_  1024
#define H_  16
#define D_  64
#define FF_ 4096
#define EPS_ 1e-5f

typedef __attribute__((ext_vector_type(16))) __bf16 v16bf;
typedef __attribute__((ext_vector_type(8)))  float  v8f;

union FragBF { v16bf v; uint32_t u[8]; };
union U32B   { uint32_t u; __bf16 b[2]; unsigned short s[2]; };

__device__ __forceinline__ __bf16 to_bf16(float f) {
    union { float f; uint32_t u; } a; a.f = f;
    uint32_t lsb = (a.u >> 16) & 1u;
    a.u += 0x7fffu + lsb;                       // round-to-nearest-even
    union { unsigned short s; __bf16 b; } c; c.s = (unsigned short)(a.u >> 16);
    return c.b;
}

// LDS byte offset of a __shared__ object = low 32 bits of its generic address
// (ISA 10.2 aperture calc).
__device__ __forceinline__ uint32_t lds_off(const void* p) {
    return (uint32_t)(uintptr_t)p;
}

// Async direct-to-LDS 16B copy (tracked by ASYNCcnt, no VGPR round trip).
__device__ __forceinline__ void async_b128(uint32_t lds, uint32_t voff, const void* base) {
    asm volatile("global_load_async_to_lds_b128 %0, %1, %2"
                 :: "v"(lds), "v"(voff), "s"(base) : "memory");
}
__device__ __forceinline__ void wait_async0() {
    asm volatile("s_wait_asynccnt 0x0" ::: "memory");
}

// Load a 16x32 bf16 WMMA fragment (A layout, or B layout when the tile is
// stored [n][k]).  Pairs along k are contiguous -> 8x 32-bit LDS loads.
// Layout (ISA 7.12.2, 16-bit A 16x32): lanes 0-15 hold K 0..7 (VGPR0-3) and
// 16..23 (VGPR4-7); lanes 16-31 hold K 8..15 and 24..31.
__device__ __forceinline__ v16bf load_frag(const __bf16* base, int ldm, int lane) {
    FragBF f;
    int m  = lane & 15;
    int hi = (lane >> 4) << 3;                  // 0 or 8
#pragma unroll
    for (int p = 0; p < 8; ++p) {
        int kl = ((p >> 2) << 4) + hi + ((p & 3) << 1);   // even, 0..30
        f.u[p] = *reinterpret_cast<const uint32_t*>(base + m * ldm + kl);
    }
    return f.v;
}

__device__ __forceinline__ float rmax16(float x) {
#pragma unroll
    for (int off = 8; off; off >>= 1) x = fmaxf(x, __shfl_xor(x, off, 32));
    return x;
}
__device__ __forceinline__ float rsum16(float x) {
#pragma unroll
    for (int off = 8; off; off >>= 1) x += __shfl_xor(x, off, 32);
    return x;
}

// ---------------------------------------------------------------------------
// f32 -> bf16 converter
// ---------------------------------------------------------------------------
__global__ void cvt_bf16_kernel(const float* __restrict__ in, __bf16* __restrict__ out, int n) {
    int i = blockIdx.x * blockDim.x + threadIdx.x;
    if (i < n) out[i] = to_bf16(in[i]);
}

// ---------------------------------------------------------------------------
// BF16 GEMM:  C[M,N] = A[M,K] * B[K,N] + bias   (optionally ReLU)
// Block = 256 thr (8 waves), tile 128x128x32, double-buffered LDS.
// A tile staged via global_load_async_to_lds_b128 (ASYNCcnt); B tile staged
// transposed through VGPRs (needed for the [n][k] fragment layout).
// Row stride 40 elems = 80 B: 16B-aligned rows for async B128 and
// bank-conflict-free fragment loads (bank = (m*20+c) mod 64, all distinct).
// ---------------------------------------------------------------------------
template<bool RELU, bool ST_F32, bool ST_BF16>
__global__ __launch_bounds__(256)
void gemm_bf16_kernel(const __bf16* __restrict__ A, const __bf16* __restrict__ B,
                      const float* __restrict__ bias,
                      float* __restrict__ Cf, __bf16* __restrict__ Cb,
                      int M, int N, int K) {
    __shared__ alignas(16) __bf16 As[2][128][40];   // [m][k]
    __shared__ alignas(16) __bf16 Bs[2][128][40];   // [n][k]  (transposed)

    const int tid  = threadIdx.x;
    const int lane = tid & 31;
    const int wid  = tid >> 5;
    const int wm   = wid & 3;        // 0..3 -> 32-row strips
    const int wn   = wid >> 2;       // 0..1 -> 64-col strips
    const int m0   = blockIdx.y * 128;
    const int n0   = blockIdx.x * 128;

    const uint32_t* Bg = reinterpret_cast<const uint32_t*>(B);
    const int N2 = N >> 1;
    const uint32_t asBase = lds_off(&As[0][0][0]);
    const uint32_t Kbytes = (uint32_t)K * 2u;

    // ---- A tile: 128 rows x 64 B = 512 x 16B chunks, 2 per thread, async
    auto stageA = [&](int kstep, int buf) {
        uint32_t base = asBase + (uint32_t)buf * (uint32_t)(128 * 40 * 2);
#pragma unroll
        for (int t = 0; t < 2; ++t) {
            int i = tid + t * 256;
            int r = i >> 2, c = i & 3;
            async_b128(base + (uint32_t)(r * 80 + c * 16),
                       (uint32_t)(m0 + r) * Kbytes + (uint32_t)(kstep * 64 + c * 16), A);
        }
    };
    // ---- B tile: 32 k-rows x 64 dwords -> regs
    auto loadBregs = [&](int kstep, uint32_t* reg) {
#pragma unroll
        for (int t = 0; t < 8; ++t) {
            int i = tid + t * 256;
            int r = i >> 6, c = i & 63;            // k = 32*kstep+r, n = n0+2c
            reg[t] = Bg[(size_t)(kstep * 32 + r) * N2 + (n0 >> 1) + c];
        }
    };
    auto storeB = [&](int buf, const uint32_t* reg) {
#pragma unroll
        for (int t = 0; t < 8; ++t) {
            int i = tid + t * 256;
            int r = i >> 6, c = i & 63;
            U32B u; u.u = reg[t];
            Bs[buf][2 * c][r]     = u.b[0];
            Bs[buf][2 * c + 1][r] = u.b[1];
        }
    };

    v8f acc[2][4];
    const v8f vzero = {0.f,0.f,0.f,0.f,0.f,0.f,0.f,0.f};
#pragma unroll
    for (int i = 0; i < 2; ++i)
#pragma unroll
        for (int j = 0; j < 4; ++j) acc[i][j] = vzero;

    const int nk = K >> 5;
    uint32_t breg[8];
    stageA(0, 0);
    loadBregs(0, breg);
    storeB(0, breg);
    wait_async0();
    __syncthreads();

    for (int k = 0; k < nk; ++k) {
        const int buf = k & 1;
        if (k + 1 < nk) {                 // prefetch next tile while computing
            stageA(k + 1, buf ^ 1);
            loadBregs(k + 1, breg);
        }

        v16bf af[2], bfr[4];
#pragma unroll
        for (int i = 0; i < 2; ++i) af[i]  = load_frag(&As[buf][wm * 32 + i * 16][0], 40, lane);
#pragma unroll
        for (int j = 0; j < 4; ++j) bfr[j] = load_frag(&Bs[buf][wn * 64 + j * 16][0], 40, lane);

#pragma unroll
        for (int i = 0; i < 2; ++i)
#pragma unroll
            for (int j = 0; j < 4; ++j)
                acc[i][j] = __builtin_amdgcn_wmma_f32_16x16x32_bf16(
                    false, af[i], false, bfr[j], (short)0, acc[i][j], false, false);

        if (k + 1 < nk) storeB(buf ^ 1, breg);
        wait_async0();
        __syncthreads();
    }

    // ---- epilogue: C layout lanes 0-15 rows v, lanes 16-31 rows v+8
    const int n_lane = lane & 15;
    const int half8  = (lane >> 4) << 3;
#pragma unroll
    for (int i = 0; i < 2; ++i) {
        int mbase = m0 + wm * 32 + i * 16 + half8;
#pragma unroll
        for (int j = 0; j < 4; ++j) {
            int col = n0 + wn * 64 + j * 16 + n_lane;
            float bv = bias[col];
#pragma unroll
            for (int v = 0; v < 8; ++v) {
                float val = acc[i][j][v] + bv;
                if (RELU) val = fmaxf(val, 0.f);
                size_t idx = (size_t)(mbase + v) * N + col;
                if (ST_F32)  Cf[idx] = val;
                if (ST_BF16) Cb[idx] = to_bf16(val);
            }
        }
    }
}

// ---------------------------------------------------------------------------
// Flash attention: grid (S/64, H), block 128 thr (4 waves, 16 queries each).
// Q,K,V are bf16 stored [s][h*64+d]; output concat heads bf16 [s][E].
// K chunk staged async-to-LDS; V chunk transposed through VGPRs.
// Row stride 72 elems = 144 B (16B-aligned, conflict-free).
// ---------------------------------------------------------------------------
__global__ __launch_bounds__(128)
void attention_kernel(const __bf16* __restrict__ Q, const __bf16* __restrict__ Km,
                      const __bf16* __restrict__ V, __bf16* __restrict__ O) {
    __shared__ alignas(16) __bf16 Ks[64][72];        // [key][d]
    __shared__ alignas(16) __bf16 Vt[64][72];        // [d][key]  (transposed)
    __shared__ alignas(16) __bf16 Ps[4][16][72];     // per-wave P staging [q][key]

    const int tid  = threadIdx.x;
    const int lane = tid & 31;
    const int w    = tid >> 5;
    const int h    = blockIdx.y;
    const int q0   = blockIdx.x * 64 + w * 16;
    const int hoff = h * D_;
    const float scale = 0.125f;          // 1/sqrt(64)

    // Q fragments (16 x 64 = two 16x32 A-frags), straight from global
    const __bf16* qbase = Q + (size_t)q0 * E_ + hoff;
    v16bf qa0 = load_frag(qbase, E_, lane);
    v16bf qa1 = load_frag(qbase + 32, E_, lane);

    const v8f vzero = {0.f,0.f,0.f,0.f,0.f,0.f,0.f,0.f};
    v8f o[4]; float mrow[8], lrow[8];
#pragma unroll
    for (int j = 0; j < 4; ++j) o[j] = vzero;
#pragma unroll
    for (int v = 0; v < 8; ++v) { mrow[v] = -1e30f; lrow[v] = 0.f; }

    const uint32_t* Vg = reinterpret_cast<const uint32_t*>(V);
    const int E2 = E_ >> 1, h2 = hoff >> 1;
    const int n_lane = lane & 15, half8 = (lane >> 4) << 3;
    const uint32_t ksBase = lds_off(&Ks[0][0]);

    for (int kb = 0; kb < S_; kb += 64) {
        // ---- K chunk: 64 rows x 128 B = 512 x 16B chunks, 4/thread, async
#pragma unroll
        for (int t = 0; t < 4; ++t) {
            int i = tid + t * 128;
            int r = i >> 3, c = i & 7;
            async_b128(ksBase + (uint32_t)(r * 144 + c * 16),
                       (uint32_t)(kb + r) * (uint32_t)(E_ * 2) +
                       (uint32_t)(hoff * 2 + c * 16), Km);
        }
        // ---- V chunk transposed via VGPRs: 64 x 32 dwords
#pragma unroll
        for (int t = 0; t < 16; ++t) {
            int i = tid + t * 128;
            int r = i >> 5, c = i & 31;
            U32B u; u.u = Vg[(size_t)(kb + r) * E2 + h2 + c];
            Vt[2 * c][r]     = u.b[0];
            Vt[2 * c + 1][r] = u.b[1];
        }
        wait_async0();
        __syncthreads();

        // ---- scores: 4 tiles of 16 keys, accumulate over d=64 (2 WMMAs)
        v8f st[4];
#pragma unroll
        for (int t = 0; t < 4; ++t) {
            v16bf kf0 = load_frag(&Ks[t * 16][0],  72, lane);
            v16bf kf1 = load_frag(&Ks[t * 16][32], 72, lane);
            v8f s = vzero;
            s = __builtin_amdgcn_wmma_f32_16x16x32_bf16(false, qa0, false, kf0,
                                                        (short)0, s, false, false);
            s = __builtin_amdgcn_wmma_f32_16x16x32_bf16(false, qa1, false, kf1,
                                                        (short)0, s, false, false);
#pragma unroll
            for (int v = 0; v < 8; ++v) s[v] *= scale;
            st[t] = s;
        }

        // ---- online softmax (rows are lane-uniform per 16-lane half)
        float alpha[8];
#pragma unroll
        for (int v = 0; v < 8; ++v) {
            float r = st[0][v];
#pragma unroll
            for (int t = 1; t < 4; ++t) r = fmaxf(r, st[t][v]);
            r = rmax16(r);
            float nm = fmaxf(mrow[v], r);
            alpha[v] = __expf(mrow[v] - nm);
            mrow[v]  = nm;
        }
#pragma unroll
        for (int t = 0; t < 4; ++t)
#pragma unroll
            for (int v = 0; v < 8; ++v)
                st[t][v] = __expf(st[t][v] - mrow[v]);
#pragma unroll
        for (int v = 0; v < 8; ++v) {
            float rs = 0.f;
#pragma unroll
            for (int t = 0; t < 4; ++t) rs += st[t][v];
            rs = rsum16(rs);
            lrow[v] = lrow[v] * alpha[v] + rs;
        }
#pragma unroll
        for (int j = 0; j < 4; ++j)
#pragma unroll
            for (int v = 0; v < 8; ++v) o[j][v] *= alpha[v];

        // ---- stage P (C layout -> A layout via LDS)
#pragma unroll
        for (int t = 0; t < 4; ++t)
#pragma unroll
            for (int v = 0; v < 8; ++v)
                Ps[w][v + half8][t * 16 + n_lane] = to_bf16(st[t][v]);
        __syncthreads();

        v16bf pa0 = load_frag(&Ps[w][0][0],  72, lane);
        v16bf pa1 = load_frag(&Ps[w][0][32], 72, lane);
#pragma unroll
        for (int j = 0; j < 4; ++j) {
            v16bf vf0 = load_frag(&Vt[j * 16][0],  72, lane);
            v16bf vf1 = load_frag(&Vt[j * 16][32], 72, lane);
            o[j] = __builtin_amdgcn_wmma_f32_16x16x32_bf16(false, pa0, false, vf0,
                                                           (short)0, o[j], false, false);
            o[j] = __builtin_amdgcn_wmma_f32_16x16x32_bf16(false, pa1, false, vf1,
                                                           (short)0, o[j], false, false);
        }
        __syncthreads();
    }

    // ---- epilogue: normalize and store concat-head output (bf16)
#pragma unroll
    for (int v = 0; v < 8; ++v) lrow[v] = 1.f / lrow[v];
#pragma unroll
    for (int j = 0; j < 4; ++j)
#pragma unroll
        for (int v = 0; v < 8; ++v) {
            size_t idx = (size_t)(q0 + v + half8) * E_ + hoff + j * 16 + n_lane;
            O[idx] = to_bf16(o[j][v] * lrow[v]);
        }
}

// ---------------------------------------------------------------------------
// y = base + LayerNorm(t; g, b); optional bf16 copy of y.  One block per row.
// ---------------------------------------------------------------------------
__global__ __launch_bounds__(256)
void add_layernorm_kernel(const float* __restrict__ base, const float* __restrict__ t,
                          const float* __restrict__ g, const float* __restrict__ b,
                          float* __restrict__ y, __bf16* __restrict__ yb) {
    __shared__ float rs[8], rs2[8];
    const int row = blockIdx.x, tid = threadIdx.x;
    const int lane = tid & 31, wave = tid >> 5;

    const float4 tv = reinterpret_cast<const float4*>(t + (size_t)row * E_)[tid];
    float s  = tv.x + tv.y + tv.z + tv.w;
    float s2 = tv.x * tv.x + tv.y * tv.y + tv.z * tv.z + tv.w * tv.w;
#pragma unroll
    for (int off = 16; off; off >>= 1) {
        s  += __shfl_xor(s,  off, 32);
        s2 += __shfl_xor(s2, off, 32);
    }
    if (lane == 0) { rs[wave] = s; rs2[wave] = s2; }
    __syncthreads();
    float S1 = 0.f, S2 = 0.f;
#pragma unroll
    for (int i = 0; i < 8; ++i) { S1 += rs[i]; S2 += rs2[i]; }
    const float mu   = S1 * (1.f / E_);
    const float var  = S2 * (1.f / E_) - mu * mu;
    const float rstd = rsqrtf(var + EPS_);

    const float4 bv = reinterpret_cast<const float4*>(base + (size_t)row * E_)[tid];
    const float4 gv = reinterpret_cast<const float4*>(g)[tid];
    const float4 bb = reinterpret_cast<const float4*>(b)[tid];
    float4 out;
    out.x = bv.x + (tv.x - mu) * rstd * gv.x + bb.x;
    out.y = bv.y + (tv.y - mu) * rstd * gv.y + bb.y;
    out.z = bv.z + (tv.z - mu) * rstd * gv.z + bb.z;
    out.w = bv.w + (tv.w - mu) * rstd * gv.w + bb.w;
    reinterpret_cast<float4*>(y + (size_t)row * E_)[tid] = out;
    if (yb) {
        __bf16* p = yb + (size_t)row * E_ + tid * 4;
        p[0] = to_bf16(out.x); p[1] = to_bf16(out.y);
        p[2] = to_bf16(out.z); p[3] = to_bf16(out.w);
    }
}

// ---------------------------------------------------------------------------
// Host-side orchestration
// ---------------------------------------------------------------------------
extern "C" void kernel_launch(void* const* d_in, const int* in_sizes, int n_in,
                              void* d_out, int out_size, void* d_ws, size_t ws_size,
                              hipStream_t stream) {
    (void)in_sizes; (void)n_in; (void)out_size; (void)ws_size;
    const float* x    = (const float*)d_in[0];
    const float* Wq   = (const float*)d_in[1];
    const float* bq   = (const float*)d_in[2];
    const float* Wk   = (const float*)d_in[3];
    const float* bk   = (const float*)d_in[4];
    const float* Wv   = (const float*)d_in[5];
    const float* bv   = (const float*)d_in[6];
    const float* Wo   = (const float*)d_in[7];
    const float* bo   = (const float*)d_in[8];
    const float* g1   = (const float*)d_in[9];
    const float* b1   = (const float*)d_in[10];
    const float* Wff1 = (const float*)d_in[11];
    const float* bff1 = (const float*)d_in[12];
    const float* Wff2 = (const float*)d_in[13];
    const float* bff2 = (const float*)d_in[14];
    const float* g2   = (const float*)d_in[15];
    const float* b2   = (const float*)d_in[16];
    float* out = (float*)d_out;

    // ---- workspace layout (bytes)
    char* w = (char*)d_ws;
    size_t off = 0;
    auto alloc = [&](size_t bytes) { char* p = w + off; off += (bytes + 255) & ~size_t(255); return p; };
    const size_t SE = (size_t)S_ * E_, EE = (size_t)E_ * E_;
    const size_t EF = (size_t)E_ * FF_, SF = (size_t)S_ * FF_;

    __bf16* xb    = (__bf16*)alloc(SE * 2);
    __bf16* wqb   = (__bf16*)alloc(EE * 2);
    __bf16* wkb   = (__bf16*)alloc(EE * 2);
    __bf16* wvb   = (__bf16*)alloc(EE * 2);
    __bf16* wob   = (__bf16*)alloc(EE * 2);
    __bf16* wf1b  = (__bf16*)alloc(EF * 2);
    __bf16* wf2b  = (__bf16*)alloc(EF * 2);
    __bf16* qb    = (__bf16*)alloc(SE * 2);
    __bf16* kb    = (__bf16*)alloc(SE * 2);
    __bf16* vbuf  = (__bf16*)alloc(SE * 2);
    __bf16* ab    = (__bf16*)alloc(SE * 2);
    float*  attn  = (float*)alloc(SE * 4);
    float*  y1    = (float*)alloc(SE * 4);
    __bf16* y1b   = (__bf16*)alloc(SE * 2);
    __bf16* f1b   = (__bf16*)alloc(SF * 2);
    float*  f2    = (float*)alloc(SE * 4);

    auto cvt = [&](const float* src, __bf16* dst, size_t n) {
        cvt_bf16_kernel<<<(unsigned)((n + 255) / 256), 256, 0, stream>>>(src, dst, (int)n);
    };
    cvt(x,    xb,   SE);
    cvt(Wq,   wqb,  EE);
    cvt(Wk,   wkb,  EE);
    cvt(Wv,   wvb,  EE);
    cvt(Wo,   wob,  EE);
    cvt(Wff1, wf1b, EF);
    cvt(Wff2, wf2b, EF);

    const dim3 gEE(E_ / 128, S_ / 128);   // N=1024
    const dim3 gEF(FF_ / 128, S_ / 128);  // N=4096

    // QKV projections -> bf16 (stored [s][h*64+d])
    gemm_bf16_kernel<false,false,true><<<gEE, 256, 0, stream>>>(xb, wqb, bq, nullptr, qb,   S_, E_, E_);
    gemm_bf16_kernel<false,false,true><<<gEE, 256, 0, stream>>>(xb, wkb, bk, nullptr, kb,   S_, E_, E_);
    gemm_bf16_kernel<false,false,true><<<gEE, 256, 0, stream>>>(xb, wvb, bv, nullptr, vbuf, S_, E_, E_);

    // Flash attention -> concat heads bf16
    attention_kernel<<<dim3(S_ / 64, H_), 128, 0, stream>>>(qb, kb, vbuf, ab);

    // Output projection -> f32
    gemm_bf16_kernel<false,true,false><<<gEE, 256, 0, stream>>>(ab, wob, bo, attn, nullptr, S_, E_, E_);

    // y1 = x + LN(attn)
    add_layernorm_kernel<<<S_, 256, 0, stream>>>(x, attn, g1, b1, y1, y1b);

    // FF1 (ReLU) -> bf16 ; FF2 -> f32
    gemm_bf16_kernel<true,false,true><<<gEF, 256, 0, stream>>>(y1b, wf1b, bff1, nullptr, f1b, S_, FF_, E_);
    gemm_bf16_kernel<false,true,false><<<gEE, 256, 0, stream>>>(f1b, wf2b, bff2, f2, nullptr, S_, E_, FF_);

    // out = y1 + LN(ff)
    add_layernorm_kernel<<<S_, 256, 0, stream>>>(y1, f2, g2, b2, out, nullptr);
}
